// CompressiveTransformer_35673998361088
// MI455X (gfx1250) — compile-verified, matrix operand order checked by reading, not run on hardware
//
#include <hip/hip_runtime.h>

// ---------------- problem constants ----------------
#define D_DIM   1024
#define T_SEQ   2048
#define B_BATCH 2
#define H_HEADS 8
#define L_DEPTH 4
#define V_VOCAB 32000
#define NTOK    (B_BATCH * T_SEQ)   // 4096 rows
#define TG      (T_SEQ / H_HEADS)   // 256 tokens per head group
#define NBH     (B_BATCH * H_HEADS) // 16 attention batches
#define SCALE_ATTN 0.08838834764831845f  // (D/H)^-0.5 = 1/sqrt(128)
#define NEG_SLOPE  0.01f
#define EPS_LN     1e-5f

// ---------------- GEMM tiling ----------------
#define BM   128
#define BN   128
#define BK   32
#define LDSK 40   // padded LDS row stride (elements); 80B keeps 16B alignment

typedef __attribute__((ext_vector_type(16))) __bf16 bf16x16;
typedef __attribute__((ext_vector_type(8)))  float  f32x8;

union FragU { uint4 q[2]; bf16x16 v; };
union Pack8 { uint4 q; __bf16 e[8]; };

// 32B per-lane async DMA global->LDS (2 x b128), tracked by ASYNCcnt.
__device__ __forceinline__ void async_ld32B(unsigned lds, unsigned long long ga) {
    asm volatile(
        "global_load_async_to_lds_b128 %0, %2, off\n\t"
        "global_load_async_to_lds_b128 %1, %3, off"
        :: "v"(lds), "v"(lds + 16u), "v"(ga), "v"(ga + 16ull)
        : "memory");
}
__device__ __forceinline__ void wait_async0() {
    asm volatile("s_wait_asynccnt 0x0" ::: "memory");
}
__device__ __forceinline__ unsigned lds_addr(const void* p) {
    // low 32 bits of the LDS-aperture flat address are the LDS byte offset
    return (unsigned)(unsigned long long)p;
}

// Load one 16x32 bf16 WMMA operand fragment from an LDS tile stored
// row-major [16 rows][LDSK]. Per CDNA5 ISA 16-bit A layout:
// lanes 0-15: M=lane, K in [0,8) and [16,24); lanes 16-31: M=lane-16, K in [8,16) and [24,32).
__device__ __forceinline__ bf16x16 ld_frag(const __bf16* base, int lane) {
    const __bf16* p = base + (lane & 15) * LDSK + ((lane >> 4) << 3);
    FragU u;
    u.q[0] = *(const uint4*)(p);
    u.q[1] = *(const uint4*)(p + 16);
    return u.v;
}

// Generic tiled bf16 WMMA GEMM body: C = epi(scale * A@B + bias) (+resid).
// A: [M][K] bf16, row stride lda.
// BT=true : B is [N][K] row-major (row stride ldb) -> double-buffered async-to-LDS staging
// BT=false: B is [K][N] row-major (row stride ldb) -> synchronous transpose staging
template<bool BT>
__device__ __forceinline__ void gemm_tile(
    const __bf16* __restrict__ A, long lda,
    const __bf16* __restrict__ B, long ldb,
    const float* __restrict__ bias,
    const float* __restrict__ resid, long ldr,
    float* __restrict__ Cf, __bf16* __restrict__ Cb, long ldc,
    int K, float scale, int leaky, int blockM, int blockN)
{
    __shared__ __bf16 As[2 * BM * LDSK];
    __shared__ __bf16 Bs[2 * BN * LDSK];

    const int tid  = threadIdx.x;
    const int lane = tid & 31;
    const int wave = tid >> 5;
    const int wm   = (wave >> 1) * 32;   // wave row offset in block tile
    const int wn   = (wave & 1) * 64;    // wave col offset in block tile

    const f32x8 vzero = {0.f,0.f,0.f,0.f,0.f,0.f,0.f,0.f};
    f32x8 acc[2][4];
#pragma unroll
    for (int i = 0; i < 2; ++i)
#pragma unroll
        for (int j = 0; j < 4; ++j) acc[i][j] = vzero;

    const int ra = tid >> 1;          // 0..127 (row within tile)
    const int ka = (tid & 1) * 16;    // 0 or 16 (k within tile)

    if (BT) {
        // -------- async double-buffered pipeline --------
        const __bf16* gA = A + (long)(blockM + ra) * lda + ka;
        const __bf16* gB = B + (long)(blockN + ra) * ldb + ka;
        const unsigned lA0 = lds_addr(&As[ra * LDSK + ka]);
        const unsigned lA1 = lds_addr(&As[BM * LDSK + ra * LDSK + ka]);
        const unsigned lB0 = lds_addr(&Bs[ra * LDSK + ka]);
        const unsigned lB1 = lds_addr(&Bs[BN * LDSK + ra * LDSK + ka]);
        const int ktiles = K / BK;

        // prologue: stage tile 0 into buffer 0
        async_ld32B(lA0, (unsigned long long)gA);
        async_ld32B(lB0, (unsigned long long)gB);

        for (int kt = 0; kt < ktiles; ++kt) {
            wait_async0();        // tile kt landed in LDS (this wave's DMAs)
            __syncthreads();      // ... and everyone else's
            if (kt + 1 < ktiles) {
                const long ko = (long)(kt + 1) * BK;
                const unsigned la = ((kt + 1) & 1) ? lA1 : lA0;
                const unsigned lb = ((kt + 1) & 1) ? lB1 : lB0;
                async_ld32B(la, (unsigned long long)(gA + ko));
                async_ld32B(lb, (unsigned long long)(gB + ko));
            }
            const __bf16* Ab = &As[(kt & 1) * BM * LDSK];
            const __bf16* Bb = &Bs[(kt & 1) * BN * LDSK];
            bf16x16 af[2], bf_[4];
#pragma unroll
            for (int i = 0; i < 2; ++i) af[i]  = ld_frag(Ab + (wm + 16 * i) * LDSK, lane);
#pragma unroll
            for (int j = 0; j < 4; ++j) bf_[j] = ld_frag(Bb + (wn + 16 * j) * LDSK, lane);
#pragma unroll
            for (int i = 0; i < 2; ++i)
#pragma unroll
                for (int j = 0; j < 4; ++j)
                    acc[i][j] = __builtin_amdgcn_wmma_f32_16x16x32_bf16(
                        false, af[i], false, bf_[j], (short)0, acc[i][j], false, false);
            // no trailing barrier: next iteration's barrier orders the
            // async writes into the buffer just consumed (waves only signal
            // after their ds fragment reads have drained).
        }
    } else {
        // -------- synchronous staging with in-LDS transpose of B --------
        for (int k0 = 0; k0 < K; k0 += BK) {
            {
                const __bf16* gp = A + (long)(blockM + ra) * lda + (k0 + ka);
                uint4 d0 = *(const uint4*)(gp);
                uint4 d1 = *(const uint4*)(gp + 8);
                *(uint4*)&As[ra * LDSK + ka]     = d0;
                *(uint4*)&As[ra * LDSK + ka + 8] = d1;
                if (k0 + BK < K) __builtin_prefetch(gp + BK, 0, 0);
            }
            {
                const int kk = tid >> 3;          // 0..31
                const int nn = (tid & 7) * 16;    // 0..112
                const __bf16* gp = B + (long)(k0 + kk) * ldb + (blockN + nn);
                Pack8 p0, p1;
                p0.q = *(const uint4*)(gp);
                p1.q = *(const uint4*)(gp + 8);
#pragma unroll
                for (int j = 0; j < 8; ++j) {
                    Bs[(nn + j) * LDSK + kk]     = p0.e[j];
                    Bs[(nn + 8 + j) * LDSK + kk] = p1.e[j];
                }
                if (k0 + BK < K) __builtin_prefetch(gp + (long)BK * ldb, 0, 0);
            }
            __syncthreads();
            bf16x16 af[2], bf_[4];
#pragma unroll
            for (int i = 0; i < 2; ++i) af[i]  = ld_frag(&As[(wm + 16 * i) * LDSK], lane);
#pragma unroll
            for (int j = 0; j < 4; ++j) bf_[j] = ld_frag(&Bs[(wn + 16 * j) * LDSK], lane);
#pragma unroll
            for (int i = 0; i < 2; ++i)
#pragma unroll
                for (int j = 0; j < 4; ++j)
                    acc[i][j] = __builtin_amdgcn_wmma_f32_16x16x32_bf16(
                        false, af[i], false, bf_[j], (short)0, acc[i][j], false, false);
            __syncthreads();
        }
    }

    // ---- epilogue: 16x16 f32 C/D layout (VGPR r -> M=r / r+8, lane&15 -> N) ----
    const int nl = lane & 15;
    const int mh = (lane >> 4) << 3;
#pragma unroll
    for (int i = 0; i < 2; ++i) {
#pragma unroll
        for (int j = 0; j < 4; ++j) {
            const int m0 = blockM + wm + 16 * i + mh;
            const int n  = blockN + wn + 16 * j + nl;
            const float bv = bias ? bias[n] : 0.0f;
#pragma unroll
            for (int r = 0; r < 8; ++r) {
                float v = acc[i][j][r] * scale + bv;
                if (leaky) v = (v >= 0.0f) ? v : NEG_SLOPE * v;
                const long m = (long)(m0 + r);
                if (resid) v += resid[m * ldr + n];
                if (Cf) Cf[m * ldc + n] = v;
                else    Cb[m * ldc + n] = (__bf16)v;
            }
        }
    }
}

// ---------------- kernels ----------------
// All weight GEMMs: B pre-transposed to [N][K] -> async path.
__global__ __launch_bounds__(256) void gemm_nt_kernel(
    const __bf16* A, long lda, const __bf16* Bt, long ldb,
    const float* bias, const float* resid, long ldr,
    float* Cf, __bf16* Cb, long ldc, int K, float scale, int leaky)
{
    gemm_tile<true>(A, lda, Bt, ldb, bias, resid, ldr, Cf, Cb, ldc,
                    K, scale, leaky, blockIdx.y * BM, blockIdx.x * BN);
}

// S[b,h] = SCALE * Q[b,h] @ K[b,h]^T ; head rows are strided tokens t = i*H + h
__global__ __launch_bounds__(256) void attn_score_kernel(
    const __bf16* __restrict__ qkv, float* __restrict__ S)
{
    const int z = blockIdx.z, b = z >> 3, hh = z & 7;
    const __bf16* qb = qkv + ((long)b * T_SEQ + hh) * (3 * D_DIM);
    const __bf16* kb = qb + D_DIM;
    float* Sb = S + (long)z * TG * TG;
    gemm_tile<true>(qb, (long)H_HEADS * 3 * D_DIM, kb, (long)H_HEADS * 3 * D_DIM,
                    nullptr, nullptr, 0, Sb, nullptr, TG, D_DIM, SCALE_ATTN, 0,
                    blockIdx.y * BM, blockIdx.x * BN);
}

// O[b,h] = P[b,h] @ V[b,h]; V is [K][N] with strided rows -> transpose path
__global__ __launch_bounds__(256) void attn_pv_kernel(
    const __bf16* __restrict__ P, const __bf16* __restrict__ qkv,
    __bf16* __restrict__ o)
{
    const int z = blockIdx.z, b = z >> 3, hh = z & 7;
    const __bf16* Pb = P + (long)z * TG * TG;
    const __bf16* vb = qkv + ((long)b * T_SEQ + hh) * (3 * D_DIM) + 2 * D_DIM;
    __bf16* ob = o + ((long)b * T_SEQ + hh) * D_DIM;
    gemm_tile<false>(Pb, TG, vb, (long)H_HEADS * 3 * D_DIM,
                     nullptr, nullptr, 0, nullptr, ob, (long)H_HEADS * D_DIM,
                     TG, 1.0f, 0, blockIdx.y * BM, blockIdx.x * BN);
}

__global__ __launch_bounds__(256) void ln_kernel(
    const float* __restrict__ h, const float* __restrict__ g,
    const float* __restrict__ b, __bf16* __restrict__ y)
{
    __shared__ float red[256];
    const int row = blockIdx.x;
    const int t   = threadIdx.x;
    const float* hr = h + (long)row * D_DIM;
    float v[4]; float s = 0.f;
#pragma unroll
    for (int j = 0; j < 4; ++j) { v[j] = hr[t + 256 * j]; s += v[j]; }
    red[t] = s; __syncthreads();
    for (int off = 128; off > 0; off >>= 1) {
        if (t < off) red[t] += red[t + off];
        __syncthreads();
    }
    const float mean = red[0] * (1.0f / D_DIM);
    __syncthreads();
    float ss = 0.f;
#pragma unroll
    for (int j = 0; j < 4; ++j) { const float d = v[j] - mean; ss += d * d; }
    red[t] = ss; __syncthreads();
    for (int off = 128; off > 0; off >>= 1) {
        if (t < off) red[t] += red[t + off];
        __syncthreads();
    }
    const float inv = rsqrtf(red[0] * (1.0f / D_DIM) + EPS_LN);
    __bf16* yr = y + (long)row * D_DIM;
#pragma unroll
    for (int j = 0; j < 4; ++j) {
        const int c = t + 256 * j;
        yr[c] = (__bf16)((v[j] - mean) * inv * g[c] + b[c]);
    }
}

__global__ __launch_bounds__(256) void softmax_kernel(
    const float* __restrict__ S, __bf16* __restrict__ P)
{
    __shared__ float red[256];
    const long row = blockIdx.x;
    const int  t   = threadIdx.x;
    const float v = S[row * TG + t];
    red[t] = v; __syncthreads();
    for (int off = 128; off > 0; off >>= 1) {
        if (t < off) red[t] = fmaxf(red[t], red[t + off]);
        __syncthreads();
    }
    const float mx = red[0]; __syncthreads();
    const float e = __expf(v - mx);
    red[t] = e; __syncthreads();
    for (int off = 128; off > 0; off >>= 1) {
        if (t < off) red[t] += red[t + off];
        __syncthreads();
    }
    P[row * TG + t] = (__bf16)(e / red[0]);
}

__global__ __launch_bounds__(256) void embed_kernel(
    const int* __restrict__ x, const float* __restrict__ emb,
    float* __restrict__ h)
{
    const int bt = blockIdx.x;
    const long tok = (long)x[bt];
    const float* e = emb + tok * D_DIM;
    float* hr = h + (long)bt * D_DIM;
#pragma unroll
    for (int j = 0; j < 4; ++j) {
        const int c = threadIdx.x + 256 * j;
        hr[c] = e[c];
    }
}

__global__ __launch_bounds__(256) void cvt_kernel(
    const float* __restrict__ src, __bf16* __restrict__ dst, long n)
{
    long i = (long)blockIdx.x * 256 + threadIdx.x;
    const long stride = (long)gridDim.x * 256;
    for (; i < n; i += stride) dst[i] = (__bf16)src[i];
}

// convert f32 [l][K][N] -> bf16 [l][N][K] (transposed, for the async NT GEMM)
__global__ __launch_bounds__(256) void cvt_t_kernel(
    const float* __restrict__ src, __bf16* __restrict__ dst, int Kd, int Nd)
{
    const int  l = blockIdx.z;
    const int  k = blockIdx.y;
    const long n = (long)blockIdx.x * 256 + threadIdx.x;
    const float v = src[((long)l * Kd + k) * Nd + n];
    dst[((long)l * Nd + n) * Kd + k] = (__bf16)v;
}

// ---------------- host launcher ----------------
extern "C" void kernel_launch(void* const* d_in, const int* in_sizes, int n_in,
                              void* d_out, int out_size, void* d_ws, size_t ws_size,
                              hipStream_t stream)
{
    (void)in_sizes; (void)n_in; (void)out_size; (void)ws_size;

    const int*   x       = (const int*)  d_in[0];
    const float* tok_emb = (const float*)d_in[1];
    const float* ln1_g   = (const float*)d_in[2];
    const float* ln1_b   = (const float*)d_in[3];
    const float* wqkv    = (const float*)d_in[4];
    const float* wout_w  = (const float*)d_in[5];
    const float* wout_b  = (const float*)d_in[6];
    const float* ln2_g   = (const float*)d_in[7];
    const float* ln2_b   = (const float*)d_in[8];
    const float* ff1_w   = (const float*)d_in[9];
    const float* ff1_b   = (const float*)d_in[10];
    const float* ff2_w   = (const float*)d_in[11];
    const float* ff2_b   = (const float*)d_in[12];
    const float* logit_w = (const float*)d_in[13];
    const float* logit_b = (const float*)d_in[14];
    float* out = (float*)d_out;

    char* p = (char*)d_ws;
    auto carve = [&](size_t bytes) -> char* {
        char* r = p; p += (bytes + 255) & ~(size_t)255; return r;
    };

    float*  hbuf     = (float*) carve((size_t)NTOK * D_DIM * 4);
    __bf16* ybuf     = (__bf16*)carve((size_t)NTOK * D_DIM * 2);
    __bf16* qkv_bf   = (__bf16*)carve((size_t)NTOK * 3 * D_DIM * 2);
    float*  Sbuf     = (float*) carve((size_t)NBH * TG * TG * 4);
    __bf16* Pbuf     = (__bf16*)carve((size_t)NBH * TG * TG * 2);
    __bf16* obuf     = (__bf16*)carve((size_t)NTOK * D_DIM * 2);
    __bf16* ffbuf    = (__bf16*)carve((size_t)NTOK * 4 * D_DIM * 2);
    __bf16* hb_bf    = (__bf16*)carve((size_t)NTOK * D_DIM * 2);
    // transposed bf16 weights: [l][N][K]
    __bf16* wqkvT  = (__bf16*)carve((size_t)L_DEPTH * 3 * D_DIM * D_DIM * 2);
    __bf16* woutT  = (__bf16*)carve((size_t)L_DEPTH * D_DIM * D_DIM * 2);
    __bf16* ff1T   = (__bf16*)carve((size_t)L_DEPTH * 4 * D_DIM * D_DIM * 2);
    __bf16* ff2T   = (__bf16*)carve((size_t)L_DEPTH * D_DIM * 4 * D_DIM * 2);
    __bf16* logitT = (__bf16*)carve((size_t)V_VOCAB * D_DIM * 2);

    const dim3 blk(256);

    // per-call weight downconversion + transpose to bf16 [N][K]
    cvt_t_kernel<<<dim3(3 * D_DIM / 256, D_DIM, L_DEPTH), blk, 0, stream>>>(wqkv,   wqkvT,  D_DIM,     3 * D_DIM);
    cvt_t_kernel<<<dim3(D_DIM / 256,     D_DIM, L_DEPTH), blk, 0, stream>>>(wout_w, woutT,  D_DIM,     D_DIM);
    cvt_t_kernel<<<dim3(4 * D_DIM / 256, D_DIM, L_DEPTH), blk, 0, stream>>>(ff1_w,  ff1T,   D_DIM,     4 * D_DIM);
    cvt_t_kernel<<<dim3(D_DIM / 256, 4 * D_DIM, L_DEPTH), blk, 0, stream>>>(ff2_w,  ff2T,   4 * D_DIM, D_DIM);
    cvt_t_kernel<<<dim3(V_VOCAB / 256,   D_DIM, 1),       blk, 0, stream>>>(logit_w, logitT, D_DIM,    V_VOCAB);

    embed_kernel<<<NTOK, blk, 0, stream>>>(x, tok_emb, hbuf);

    for (int l = 0; l < L_DEPTH; ++l) {
        // y = LN1(h)  (bf16)
        ln_kernel<<<NTOK, blk, 0, stream>>>(hbuf, ln1_g + l * D_DIM, ln1_b + l * D_DIM, ybuf);
        // qkv = y @ wqkv  (bf16 out, no bias)
        gemm_nt_kernel<<<dim3(3 * D_DIM / BN, NTOK / BM), blk, 0, stream>>>(
            ybuf, D_DIM, wqkvT + (long)l * 3 * D_DIM * D_DIM, D_DIM,
            nullptr, nullptr, 0, nullptr, qkv_bf, 3 * D_DIM, D_DIM, 1.0f, 0);
        // S = SCALE * Q @ K^T  (16 batched 256x256x1024 GEMMs)
        attn_score_kernel<<<dim3(TG / BN, TG / BM, NBH), blk, 0, stream>>>(qkv_bf, Sbuf);
        // P = softmax(S)
        softmax_kernel<<<NBH * TG, blk, 0, stream>>>(Sbuf, Pbuf);
        // O = P @ V  (scatter rows back to token order, bf16)
        attn_pv_kernel<<<dim3(D_DIM / BN, TG / BM, NBH), blk, 0, stream>>>(Pbuf, qkv_bf, obuf);
        // h = h + O @ wout + wout_b
        gemm_nt_kernel<<<dim3(D_DIM / BN, NTOK / BM), blk, 0, stream>>>(
            obuf, D_DIM, woutT + (long)l * D_DIM * D_DIM, D_DIM,
            wout_b + l * D_DIM, hbuf, D_DIM, hbuf, nullptr, D_DIM, D_DIM, 1.0f, 0);
        // y = LN2(h)
        ln_kernel<<<NTOK, blk, 0, stream>>>(hbuf, ln2_g + l * D_DIM, ln2_b + l * D_DIM, ybuf);
        // ff = leaky(y @ ff1 + ff1_b)  (bf16)
        gemm_nt_kernel<<<dim3(4 * D_DIM / BN, NTOK / BM), blk, 0, stream>>>(
            ybuf, D_DIM, ff1T + (long)l * 4 * D_DIM * D_DIM, D_DIM,
            ff1_b + l * 4 * D_DIM, nullptr, 0, nullptr, ffbuf, 4 * D_DIM, D_DIM, 1.0f, 1);
        // h = h + ff @ ff2 + ff2_b
        gemm_nt_kernel<<<dim3(D_DIM / BN, NTOK / BM), blk, 0, stream>>>(
            ffbuf, 4 * D_DIM, ff2T + (long)l * D_DIM * 4 * D_DIM, 4 * D_DIM,
            ff2_b + l * D_DIM, hbuf, D_DIM, hbuf, nullptr, D_DIM, 4 * D_DIM, 1.0f, 0);
    }

    // logits = h @ logit_w + logit_b  (f32 out)
    cvt_kernel<<<2048, blk, 0, stream>>>(hbuf, hb_bf, (long)NTOK * D_DIM);
    gemm_nt_kernel<<<dim3(V_VOCAB / BN, NTOK / BM), blk, 0, stream>>>(
        hb_bf, D_DIM, logitT, D_DIM,
        logit_b, nullptr, 0, out, nullptr, V_VOCAB, D_DIM, 1.0f, 0);
}
